// Attention_11458972746115
// MI455X (gfx1250) — matrix operand order repaired and checked by reading, not compiled, not run on hardware
//
#include <hip/hip_runtime.h>
#include <hip/hip_bf16.h>

typedef __attribute__((ext_vector_type(16))) __bf16 bf16x16;
typedef __attribute__((ext_vector_type(8)))  float   f32x8;

#define S_LEN  2048
#define D_DIM  1024
#define H_NUM  16
#define DH_DIM 64

#ifndef __has_builtin
#define __has_builtin(x) 0
#endif

#if __has_builtin(__builtin_amdgcn_global_load_async_to_lds_b32)
#define HAVE_ASYNC_LDS 1
#else
#define HAVE_ASYNC_LDS 0
#endif

#if __has_builtin(__builtin_amdgcn_s_wait_asynccnt)
#define ASYNC_WAIT() __builtin_amdgcn_s_wait_asynccnt(0)
#else
#define ASYNC_WAIT() asm volatile("s_wait_asynccnt 0" ::: "memory")
#endif

#define WMMA_BF16(a, b, c) \
  __builtin_amdgcn_wmma_f32_16x16x32_bf16(false, (a), false, (b), (short)0, (c), false, false)

// A-fragment (16x32 bf16): lane holds row m0+(l&15); lanes<16 K {0..7,16..23},
// lanes>=16 K {8..15,24..31}.  Two 16-byte contiguous chunks per lane.
static __device__ inline bf16x16 load_afrag(const __bf16* __restrict__ base, int ld,
                                            int m0, int koff, int lane) {
  const __bf16* p = base + (size_t)(m0 + (lane & 15)) * ld + koff + ((lane >> 4) * 8);
  bf16x16 a;
  ((uint4*)&a)[0] = ((const uint4*)p)[0];
  ((uint4*)&a)[1] = *(const uint4*)(p + 16);
  return a;
}

// B-fragment (32x16): B[k][n] = base[n][k] with row n contiguous along k.
// lane holds col n0+(l&15); lanes<16 K {0..7,16..23}, lanes>=16 K {8..15,24..31}. Two 16-byte chunks.
static __device__ inline bf16x16 load_bfrag(const __bf16* __restrict__ base, int ld,
                                            int n0, int koff, int lane) {
  const __bf16* p = base + (size_t)(n0 + (lane & 15)) * ld + koff + ((lane >> 4) * 8);
  bf16x16 b;
  ((uint4*)&b)[0] = ((const uint4*)p)[0];
  ((uint4*)&b)[1] = *(const uint4*)(p + 16);
  return b;
}

// ---------------- conversion kernels ----------------

// fp32 -> bf16, 4 elements per thread
__global__ __launch_bounds__(256) void cvt_bf16(const float* __restrict__ src,
                                                __bf16* __restrict__ dst, int n) {
  const int base = (blockIdx.x * blockDim.x + threadIdx.x) * 4;
  if (base >= n) return;
  const float4 f = *(const float4*)(src + base);
  dst[base + 0] = (__bf16)f.x;
  dst[base + 1] = (__bf16)f.y;
  dst[base + 2] = (__bf16)f.z;
  dst[base + 3] = (__bf16)f.w;
}

// V[S,D] fp32 -> Vt[D,S] bf16 (transpose so P@V B-fragments are contiguous)
__global__ __launch_bounds__(256) void cvt_transpose(const float* __restrict__ v,
                                                     __bf16* __restrict__ vt) {
  const int idx = blockIdx.x * blockDim.x + threadIdx.x;  // S*D threads
  const int d = idx & (D_DIM - 1);
  const int s = idx >> 10;
  vt[(size_t)d * S_LEN + s] = (__bf16)v[(size_t)s * D_DIM + d];
}

// RoPE on fp32 q/k, emitting bf16 copies for the attention kernel.
__global__ __launch_bounds__(256) void rope_cvt(const float* __restrict__ qf,
                                                const float* __restrict__ kf,
                                                const float* __restrict__ cosT,
                                                const float* __restrict__ sinT,
                                                __bf16* __restrict__ Qb,
                                                __bf16* __restrict__ Kb) {
  const int idx = blockIdx.x * blockDim.x + threadIdx.x;  // S*H*32 threads
  const int i = idx & 31;
  const int h = (idx >> 5) & (H_NUM - 1);
  const int s = idx >> 9;
  const float c  = cosT[s * (DH_DIM / 2) + i];
  const float sn = sinT[s * (DH_DIM / 2) + i];
  const size_t base = (size_t)s * D_DIM + h * DH_DIM + 2 * i;
  float a = qf[base], b = qf[base + 1];
  Qb[base]     = (__bf16)(a * c - b * sn);
  Qb[base + 1] = (__bf16)(a * sn + b * c);
  a = kf[base]; b = kf[base + 1];
  Kb[base]     = (__bf16)(a * c - b * sn);
  Kb[base + 1] = (__bf16)(a * sn + b * c);
}

// ---------------- GEMM: Y = X @ W^T (bf16 in, f32 out) ----------------
// Block = 8 waves; wave computes a 16x64 strip (4 WMMA tiles) so one A-fragment
// feeds 4 matrix ops.  The 16x32 A tile (shared by all waves) is staged in LDS
// via async DMA.  grid = (N/512, M/16).
__global__ __launch_bounds__(256) void gemm_bf16(const __bf16* __restrict__ X,
                                                 const __bf16* __restrict__ W,
                                                 float* __restrict__ Y,
                                                 int M, int N, int K) {
  __shared__ __align__(16) __bf16 aTile[16 * 32];
  const int tid    = threadIdx.x;
  const int lane   = tid & 31;
  const int wave   = tid >> 5;
  const int half   = lane >> 4;
  const int lanelo = lane & 15;
  const int n0 = blockIdx.x * 512 + wave * 64;
  const int m0 = blockIdx.y * 16;
  const int arow = tid >> 4;        // 0..15 : A-tile row
  const int acol = (tid & 15) * 2;  // bf16 column offset (4 B per thread)

  f32x8 acc0 = {}, acc1 = {}, acc2 = {}, acc3 = {};

  for (int kb = 0; kb < K; kb += 32) {
    __syncthreads();  // WAR: previous iteration's LDS reads complete
    {
      const __bf16* gp = X + (size_t)(m0 + arow) * K + kb + acol;
      __bf16* lp = &aTile[arow * 32 + acol];
#if HAVE_ASYNC_LDS
      __builtin_amdgcn_global_load_async_to_lds_b32(
          (__attribute__((address_space(1))) int*)gp,
          (__attribute__((address_space(3))) int*)lp, 0, 0);
      ASYNC_WAIT();
#else
      *(unsigned int*)lp = *(const unsigned int*)gp;
#endif
    }
    __syncthreads();

    bf16x16 a;
    {
      const __bf16* ap = &aTile[lanelo * 32 + half * 8];
      ((uint4*)&a)[0] = ((const uint4*)ap)[0];
      ((uint4*)&a)[1] = *(const uint4*)(ap + 16);
    }
    {
      bf16x16 b = load_bfrag(W, K, n0 + 0,  kb, lane);
      acc0 = WMMA_BF16(a, b, acc0);
      b = load_bfrag(W, K, n0 + 16, kb, lane);
      acc1 = WMMA_BF16(a, b, acc1);
      b = load_bfrag(W, K, n0 + 32, kb, lane);
      acc2 = WMMA_BF16(a, b, acc2);
      b = load_bfrag(W, K, n0 + 48, kb, lane);
      acc3 = WMMA_BF16(a, b, acc3);
    }
  }

  const int nc = n0 + lanelo;
#pragma unroll
  for (int j = 0; j < 8; ++j) {
    float* py = Y + (size_t)(m0 + half * 8 + j) * N;
    py[nc +  0] = acc0[j];
    py[nc + 16] = acc1[j];
    py[nc + 32] = acc2[j];
    py[nc + 48] = acc3[j];
  }
}

// ---------------- Flash attention with segment mask ----------------
// One wave per (16-query tile, head).  grid = (S/16/8, H), block = 256.
__global__ __launch_bounds__(256) void flash_attn(const __bf16* __restrict__ Qb,
                                                  const __bf16* __restrict__ Kb,
                                                  const __bf16* __restrict__ Vt,
                                                  const int* __restrict__ seg,
                                                  float* __restrict__ o) {
  __shared__ __align__(16) __bf16 lds_p[8][16 * 32];  // per-wave P staging
  const int lane   = threadIdx.x & 31;
  const int wave   = threadIdx.x >> 5;
  const int half   = lane >> 4;
  const int lanelo = lane & 15;
  const int h  = blockIdx.y;
  const int q0 = (blockIdx.x * 8 + wave) * 16;
  const int hc = h * DH_DIM;

  const bf16x16 qa0 = load_afrag(Qb, D_DIM, q0, hc + 0,  lane);
  const bf16x16 qa1 = load_afrag(Qb, D_DIM, q0, hc + 32, lane);

  int segq[8];
#pragma unroll
  for (int j = 0; j < 8; ++j) segq[j] = seg[q0 + half * 8 + j];

  f32x8 acc0 = {}, acc1 = {}, acc2 = {}, acc3 = {};
  float mrow[8], lrow[8];
#pragma unroll
  for (int j = 0; j < 8; ++j) { mrow[j] = -1e30f; lrow[j] = 0.0f; }

  for (int kb = 0; kb < S_LEN; kb += 32) {
    // Scores for two 16-key subtiles (4 WMMAs)
    f32x8 s0 = {}, s1 = {};
    s0 = WMMA_BF16(qa0, load_bfrag(Kb, D_DIM, kb,      hc + 0,  lane), s0);
    s0 = WMMA_BF16(qa1, load_bfrag(Kb, D_DIM, kb,      hc + 32, lane), s0);
    s1 = WMMA_BF16(qa0, load_bfrag(Kb, D_DIM, kb + 16, hc + 0,  lane), s1);
    s1 = WMMA_BF16(qa1, load_bfrag(Kb, D_DIM, kb + 16, hc + 32, lane), s1);

    const int sk0 = seg[kb + lanelo];
    const int sk1 = seg[kb + 16 + lanelo];

    float fac[8];
#pragma unroll
    for (int j = 0; j < 8; ++j) {
      float x0 = (sk0 == segq[j]) ? s0[j] * 0.125f : -1e9f;   // 1/sqrt(64)
      float x1 = (sk1 == segq[j]) ? s1[j] * 0.125f : -1e9f;
      float rm = fmaxf(x0, x1);                                // row max across half
      rm = fmaxf(rm, __shfl_xor(rm, 1));
      rm = fmaxf(rm, __shfl_xor(rm, 2));
      rm = fmaxf(rm, __shfl_xor(rm, 4));
      rm = fmaxf(rm, __shfl_xor(rm, 8));
      const float mn = fmaxf(mrow[j], rm);
      const float p0 = __expf(x0 - mn);
      const float p1 = __expf(x1 - mn);
      float rs = p0 + p1;
      rs += __shfl_xor(rs, 1);
      rs += __shfl_xor(rs, 2);
      rs += __shfl_xor(rs, 4);
      rs += __shfl_xor(rs, 8);
      const float f = __expf(mrow[j] - mn);
      lrow[j] = lrow[j] * f + rs;
      mrow[j] = mn;
      fac[j]  = f;
      lds_p[wave][(half * 8 + j) * 32 + lanelo]      = (__bf16)p0;
      lds_p[wave][(half * 8 + j) * 32 + 16 + lanelo] = (__bf16)p1;
    }
#pragma unroll
    for (int j = 0; j < 8; ++j) {
      acc0[j] *= fac[j]; acc1[j] *= fac[j];
      acc2[j] *= fac[j]; acc3[j] *= fac[j];
    }
    asm volatile("s_wait_dscnt 0" ::: "memory");

    // Reload P in A-fragment layout (same wave; DS ops in-order)
    bf16x16 pa;
    {
      const __bf16* pp = &lds_p[wave][lanelo * 32 + half * 8];
      ((uint4*)&pa)[0] = ((const uint4*)pp)[0];
      ((uint4*)&pa)[1] = *(const uint4*)(pp + 16);
    }
    // O += P(16x32) @ V(32x64); Vt is [D,S] so B-fragments are contiguous
    acc0 = WMMA_BF16(pa, load_bfrag(Vt, S_LEN, hc + 0,  kb, lane), acc0);
    acc1 = WMMA_BF16(pa, load_bfrag(Vt, S_LEN, hc + 16, kb, lane), acc1);
    acc2 = WMMA_BF16(pa, load_bfrag(Vt, S_LEN, hc + 32, kb, lane), acc2);
    acc3 = WMMA_BF16(pa, load_bfrag(Vt, S_LEN, hc + 48, kb, lane), acc3);
  }

#pragma unroll
  for (int j = 0; j < 8; ++j) {
    const float inv = 1.0f / lrow[j];
    float* po = o + (size_t)(q0 + half * 8 + j) * D_DIM + hc;
    po[lanelo]      = acc0[j] * inv;
    po[16 + lanelo] = acc1[j] * inv;
    po[32 + lanelo] = acc2[j] * inv;
    po[48 + lanelo] = acc3[j] * inv;
  }
}

extern "C" void kernel_launch(void* const* d_in, const int* in_sizes, int n_in,
                              void* d_out, int out_size, void* d_ws, size_t ws_size,
                              hipStream_t stream) {
  const float* x   = (const float*)d_in[0];
  const int*   seg = (const int*)d_in[1];
  const float* fc  = (const float*)d_in[2];
  const float* fs  = (const float*)d_in[3];
  const float* wq  = (const float*)d_in[4];
  const float* wk  = (const float*)d_in[5];
  const float* wv  = (const float*)d_in[6];
  const float* wo  = (const float*)d_in[7];
  float* out = (float*)d_out;

  const size_t SD = (size_t)S_LEN * D_DIM;   // 2M elements
  const size_t DD = (size_t)D_DIM * D_DIM;   // 1M elements

  char* w = (char*)d_ws;
  __bf16* xb  = (__bf16*)w; w += SD * 2;
  __bf16* wqb = (__bf16*)w; w += DD * 2;
  __bf16* wkb = (__bf16*)w; w += DD * 2;
  __bf16* wvb = (__bf16*)w; w += DD * 2;
  __bf16* wob = (__bf16*)w; w += DD * 2;
  float*  qf  = (float*)w;  w += SD * 4;
  float*  kf  = (float*)w;  w += SD * 4;
  float*  vf  = (float*)w;  w += SD * 4;
  __bf16* Qb  = (__bf16*)w; w += SD * 2;
  __bf16* Kb  = (__bf16*)w; w += SD * 2;
  __bf16* Vt  = (__bf16*)w; w += SD * 2;
  float*  ao  = (float*)w;  w += SD * 4;
  __bf16* aob = (__bf16*)w; w += SD * 2;

  // 1) one-pass fp32 -> bf16 conversions
  cvt_bf16<<<SD / 4 / 256, 256, 0, stream>>>(x,  xb,  (int)SD);
  cvt_bf16<<<DD / 4 / 256, 256, 0, stream>>>(wq, wqb, (int)DD);
  cvt_bf16<<<DD / 4 / 256, 256, 0, stream>>>(wk, wkb, (int)DD);
  cvt_bf16<<<DD / 4 / 256, 256, 0, stream>>>(wv, wvb, (int)DD);
  cvt_bf16<<<DD / 4 / 256, 256, 0, stream>>>(wo, wob, (int)DD);

  // 2) QKV projections
  const dim3 gg(D_DIM / 512, S_LEN / 16);
  gemm_bf16<<<gg, 256, 0, stream>>>(xb, wqb, qf, S_LEN, D_DIM, D_DIM);
  gemm_bf16<<<gg, 256, 0, stream>>>(xb, wkb, kf, S_LEN, D_DIM, D_DIM);
  gemm_bf16<<<gg, 256, 0, stream>>>(xb, wvb, vf, S_LEN, D_DIM, D_DIM);

  // 3) RoPE (+ bf16), V transpose (+ bf16)
  rope_cvt<<<(S_LEN * H_NUM * (DH_DIM / 2)) / 256, 256, 0, stream>>>(qf, kf, fc, fs, Qb, Kb);
  cvt_transpose<<<SD / 256, 256, 0, stream>>>(vf, Vt);

  // 4) attention
  flash_attn<<<dim3(S_LEN / 16 / 8, H_NUM), 256, 0, stream>>>(Qb, Kb, Vt, seg, ao);

  // 5) output projection
  cvt_bf16<<<SD / 4 / 256, 256, 0, stream>>>(ao, aob, (int)SD);
  gemm_bf16<<<gg, 256, 0, stream>>>(aob, wob, out, S_LEN, D_DIM, D_DIM);
}
